// LatticeAttentionHead_35467839930937
// MI455X (gfx1250) — matrix operand order repaired and checked by reading, not compiled, not run on hardware
//
#include <hip/hip_runtime.h>
#include <math.h>

// Problem dims (fixed by the reference)
#define B_  4
#define L_  2048
#define D_  1024
#define DK_ 128

typedef __bf16 bf16_t;
typedef __attribute__((ext_vector_type(16))) __bf16 v16bf;
typedef __attribute__((ext_vector_type(8)))  __bf16 v8bf;
typedef __attribute__((ext_vector_type(8)))  float  v8f;
typedef __attribute__((ext_vector_type(4)))  float  v4f;

// ---------------------------------------------------------------------------
// WMMA fragment helpers (layouts per CDNA5 ISA, wave32)
//
// A (16x32 bf16, MxK): lane m = lane&15 is the row, h = lane>>4 selects K half.
//   elems 0..7  -> K = h*8 + e          (contiguous 8)
//   elems 8..15 -> K = 16 + h*8 + (e-8) (contiguous 8)
// B (32x16 bf16, KxN): lane n = lane&15 is the column, h = lane>>4.
//   elems 0..7  -> K = h*8 + e          (contiguous 8)
//   elems 8..15 -> K = 16 + h*8 + (e-8) (contiguous 8)
// C/D (16x16 f32): lane n = lane&15 col, rows M = (lane>>4)*8 + r for r=0..7.
// ---------------------------------------------------------------------------

static __device__ inline v16bf load_a_bf16(const bf16_t* rowbase, int h) {
  v8bf lo = *(const v8bf*)(rowbase + h * 8);
  v8bf hi = *(const v8bf*)(rowbase + 16 + h * 8);
  v16bf r;
#pragma unroll
  for (int i = 0; i < 8; ++i) { r[i] = lo[i]; r[i + 8] = hi[i]; }
  return r;
}

static __device__ inline v16bf load_a_f32(const float* rowbase, int h) {
  v8f lo = *(const v8f*)(rowbase + h * 8);
  v8f hi = *(const v8f*)(rowbase + 16 + h * 8);
  v16bf r;
#pragma unroll
  for (int i = 0; i < 8; ++i) {
    r[i]     = (bf16_t)lo[i];
    r[i + 8] = (bf16_t)hi[i];
  }
  return r;
}

// Same fragment but sourced from (16-byte aligned) LDS rows of f32.
static __device__ inline v16bf lds_a_from_f32(const float* rowbase, int h) {
  v4f a0 = *(const v4f*)(rowbase + h * 8);
  v4f a1 = *(const v4f*)(rowbase + h * 8 + 4);
  v4f b0 = *(const v4f*)(rowbase + 16 + h * 8);
  v4f b1 = *(const v4f*)(rowbase + 16 + h * 8 + 4);
  v16bf r;
#pragma unroll
  for (int i = 0; i < 4; ++i) {
    r[i]      = (bf16_t)a0[i];
    r[i + 4]  = (bf16_t)a1[i];
    r[i + 8]  = (bf16_t)b0[i];
    r[i + 12] = (bf16_t)b1[i];
  }
  return r;
}

static __device__ inline v16bf load_b_bf16(const bf16_t* colbase, int h) {
  v8bf lo = *(const v8bf*)(colbase + h * 8);
  v8bf hi = *(const v8bf*)(colbase + 16 + h * 8);
  v16bf r;
#pragma unroll
  for (int i = 0; i < 8; ++i) { r[i] = lo[i]; r[i + 8] = hi[i]; }
  return r;
}

static __device__ inline v8f wmma_bf16(v16bf a, v16bf b, v8f c) {
  return __builtin_amdgcn_wmma_f32_16x16x32_bf16(
      /*neg_a=*/false, a, /*neg_b=*/false, b,
      /*c_mod=*/(short)0, c, /*reuse_a=*/false, /*reuse_b=*/false);
}

// ---------------------------------------------------------------------------
// gfx1250 async global->LDS copy (CDNA5 ISA).
// Each lane copies 16 bytes; one ASYNCcnt increment per wave per call.
// Generic LDS pointers on amdgcn carry the DS byte address in the low 32 bits.
// ---------------------------------------------------------------------------
static __device__ inline void async_copy_b128(const float* gptr, float* lptr) {
  unsigned int  laddr = (unsigned int)(unsigned long long)(uintptr_t)lptr;
  unsigned long long gaddr = (unsigned long long)(uintptr_t)gptr;
  asm volatile("global_load_async_to_lds_b128 %0, %1, off"
               :
               : "v"(laddr), "v"(gaddr)
               : "memory");
}

static __device__ inline void wait_asynccnt_le1() {
  asm volatile("s_wait_asynccnt 0x1" ::: "memory");
}
static __device__ inline void wait_asynccnt_0() {
  asm volatile("s_wait_asynccnt 0x0" ::: "memory");
}

// ---------------------------------------------------------------------------
// Kernel 1: W [D, DK] f32 (row-major, x@W) -> Wt [DK, D] bf16 (transposed)
// ---------------------------------------------------------------------------
__global__ void wtrans_kernel(const float* __restrict__ W, bf16_t* __restrict__ Wt) {
  int idx = blockIdx.x * 256 + threadIdx.x;      // idx over D*DK, coalesced read
  int k = idx / DK_;
  int n = idx % DK_;
  Wt[(size_t)n * D_ + k] = (bf16_t)W[idx];
}

// ---------------------------------------------------------------------------
// Kernel 2: projection  Out = bf16( X[B*L, D] @ W )   (W given as Wt[DK, D])
// transposed==0 : Out[B*L, DK] row-major (Q, K)
// transposed==1 : Out[b, DK, L]          (V, so attn@V B-frags are contiguous)
// grid.x = (B*L)/16, block = 256 (8 waves; wave w owns column tile w)
// ---------------------------------------------------------------------------
__global__ void proj_kernel(const float* __restrict__ X,
                            const bf16_t* __restrict__ Wt,
                            bf16_t* __restrict__ Out, int transposed) {
  const int m0   = blockIdx.x * 16;
  const int wave = threadIdx.x >> 5;
  const int lane = threadIdx.x & 31;
  const int h = lane >> 4;
  const int n = lane & 15;   // A row index AND B/D column index (same bits)

  const float*  xrow  = X  + (size_t)(m0 + n) * D_;
  const bf16_t* wtcol = Wt + (size_t)(wave * 16 + n) * D_;

  v8f acc = {};
#pragma unroll 4
  for (int ks = 0; ks < D_ / 32; ++ks) {
    v16bf a = load_a_f32(xrow + ks * 32, h);
    v16bf b = load_b_bf16(wtcol + ks * 32, h);
    acc = wmma_bf16(a, b, acc);
  }

#pragma unroll
  for (int r = 0; r < 8; ++r) {
    int row = m0 + h * 8 + r;
    int col = wave * 16 + n;
    if (!transposed) {
      Out[(size_t)row * DK_ + col] = (bf16_t)acc[r];
    } else {
      int b = row / L_;
      int l = row % L_;
      Out[((size_t)b * DK_ + col) * L_ + l] = (bf16_t)acc[r];
    }
  }
}

// ---------------------------------------------------------------------------
// Kernel 3: raw scores = Q Kt / sqrt(DK) + bias, masked; written f32 to attn
// grid = (L/16, B), block = 256. Waves stride over 128 column tiles.
// ---------------------------------------------------------------------------
__global__ void scores_kernel(const bf16_t* __restrict__ Qb,
                              const bf16_t* __restrict__ Kb,
                              const int*    __restrict__ mask,
                              const float*  __restrict__ nph,
                              const float*  __restrict__ nw,
                              const float*  __restrict__ cs,
                              float* __restrict__ attn) {
  const int b    = blockIdx.y;
  const int m0   = blockIdx.x * 16;
  const int wave = threadIdx.x >> 5;
  const int lane = threadIdx.x & 31;
  const int h = lane >> 4;
  const int n = lane & 15;

  // Kuramoto phase bias (scalar; phase buffer starts at 0, intrinsic freq 1)
  float coupling = 0.0f;
#pragma unroll
  for (int i = 0; i < 4; ++i) coupling += nw[i] * __sinf(nph[i]);
  float phase = fmodf(0.01f * (1.0f + cs[0] * coupling), 6.283185307179586f);
  const float bias = 0.1f * __cosf(phase);
  const float inv_scale = 0.08838834764831845f;  // 1/sqrt(128)

  // Preload the 4 A-fragments for this 16-row Q strip; reuse across col tiles.
  const bf16_t* qbase = Qb + ((size_t)b * L_ + m0 + n) * DK_;
  v16bf aq[4];
#pragma unroll
  for (int ks = 0; ks < 4; ++ks) aq[ks] = load_a_bf16(qbase + ks * 32, h);

  for (int ct = wave; ct < L_ / 16; ct += 8) {
    const bf16_t* kbase = Kb + ((size_t)b * L_ + ct * 16 + n) * DK_;
    v8f acc = {};
#pragma unroll
    for (int ks = 0; ks < 4; ++ks)
      acc = wmma_bf16(aq[ks], load_b_bf16(kbase + ks * 32, h), acc);

    const int kcol = ct * 16 + n;
#pragma unroll
    for (int r = 0; r < 8; ++r) {
      int q = m0 + h * 8 + r;
      float s = acc[r] * inv_scale + bias;
      if (mask[(size_t)q * L_ + kcol] == 0) s = -10000.0f;
      attn[((size_t)b * L_ + q) * L_ + kcol] = s;
    }
  }
}

// ---------------------------------------------------------------------------
// Kernel 4: in-place row softmax over 2048 cols. One block per row,
// 8 regs/thread, shfl_xor wave32 reduce + tiny LDS cross-wave reduce.
// ---------------------------------------------------------------------------
__global__ void softmax_kernel(float* __restrict__ attn) {
  float* p = attn + (size_t)blockIdx.x * L_;
  const int tid = threadIdx.x;

  float x[8];
#pragma unroll
  for (int i = 0; i < 8; ++i) x[i] = p[tid + i * 256];

  float mx = x[0];
#pragma unroll
  for (int i = 1; i < 8; ++i) mx = fmaxf(mx, x[i]);
#pragma unroll
  for (int off = 16; off > 0; off >>= 1) mx = fmaxf(mx, __shfl_xor(mx, off, 32));

  __shared__ float redmax[8];
  __shared__ float redsum[8];
  if ((tid & 31) == 0) redmax[tid >> 5] = mx;
  __syncthreads();
  mx = redmax[0];
#pragma unroll
  for (int w = 1; w < 8; ++w) mx = fmaxf(mx, redmax[w]);

  float s = 0.0f;
#pragma unroll
  for (int i = 0; i < 8; ++i) { x[i] = __expf(x[i] - mx); s += x[i]; }
#pragma unroll
  for (int off = 16; off > 0; off >>= 1) s += __shfl_xor(s, off, 32);
  if ((tid & 31) == 0) redsum[tid >> 5] = s;
  __syncthreads();
  s = 0.0f;
#pragma unroll
  for (int w = 0; w < 8; ++w) s += redsum[w];

  const float inv = 1.0f / s;
#pragma unroll
  for (int i = 0; i < 8; ++i) p[tid + i * 256] = x[i] * inv;
}

// ---------------------------------------------------------------------------
// Kernel 5: out = attn @ V.
// attn rows staged once per block via async global->LDS DMA (double-buffered,
// ASYNCcnt-tracked), removing the 8x per-wave redundant reads of attn.
// LDS rows padded to 68 f32 (272 B) so per-lane ds_load_b128 fragment reads
// start at bank 4*lane -> conflict-free across the 16 lanes of each half.
// grid = (L/16, B), block = 256; wave w owns output column tile w, K = 2048.
// ---------------------------------------------------------------------------
#define KC_     64      // K-chunk per stage: 16 rows x 64 f32 = 4 KB = 256 lanes x 16 B
#define LDSROW_ 68      // padded row length in f32

__global__ void av_kernel(const float* __restrict__ attn,
                          const bf16_t* __restrict__ Vt,
                          float* __restrict__ out) {
  __shared__ __attribute__((aligned(16))) float abuf[2][16 * LDSROW_];

  const int b    = blockIdx.y;
  const int m0   = blockIdx.x * 16;
  const int tid  = threadIdx.x;
  const int wave = tid >> 5;
  const int lane = tid & 31;
  const int h = lane >> 4;
  const int n = lane & 15;

  // Staging coordinates: thread copies 16 B (4 f32) of the 16x64 chunk.
  const int srow  = tid >> 4;   // 0..15
  const int squad = tid & 15;   // 0..15
  const float* gsrc = attn + ((size_t)b * L_ + m0 + srow) * L_ + squad * 4;
  float* ldst[2] = { &abuf[0][srow * LDSROW_ + squad * 4],
                     &abuf[1][srow * LDSROW_ + squad * 4] };

  const bf16_t* vcol = Vt + ((size_t)b * DK_ + wave * 16 + n) * L_;

  v8f acc = {};

  async_copy_b128(gsrc, ldst[0]);                    // preload chunk 0

  for (int c = 0; c < L_ / KC_; ++c) {
    const int cur = c & 1;
    if (c + 1 < L_ / KC_) {
      async_copy_b128(gsrc + (size_t)(c + 1) * KC_, ldst[cur ^ 1]);
      wait_asynccnt_le1();     // my chunk-c op (older) has landed
    } else {
      wait_asynccnt_0();
    }
    __syncthreads();           // all waves' portions of chunk c visible

    const float* arow = &abuf[cur][n * LDSROW_];
#pragma unroll
    for (int kl = 0; kl < KC_ / 32; ++kl) {
      v16bf a = lds_a_from_f32(arow + kl * 32, h);
      v16bf v = load_b_bf16(vcol + c * KC_ + kl * 32, h);
      acc = wmma_bf16(a, v, acc);
    }
    __syncthreads();           // buffer `cur` safe to overwrite at c+2's issue
  }

#pragma unroll
  for (int r = 0; r < 8; ++r) {
    int q = m0 + h * 8 + r;
    out[((size_t)b * L_ + q) * DK_ + wave * 16 + n] = acc[r];
  }
}

// ---------------------------------------------------------------------------
extern "C" void kernel_launch(void* const* d_in, const int* in_sizes, int n_in,
                              void* d_out, int out_size, void* d_ws, size_t ws_size,
                              hipStream_t stream) {
  (void)in_sizes; (void)n_in; (void)out_size; (void)ws_size;

  const float* query = (const float*)d_in[0];
  const float* key   = (const float*)d_in[1];
  const float* value = (const float*)d_in[2];
  const float* nph   = (const float*)d_in[3];
  const float* nw    = (const float*)d_in[4];
  const int*   mask  = (const int*)d_in[5];
  const float* Wq    = (const float*)d_in[6];
  const float* Wk    = (const float*)d_in[7];
  const float* Wv    = (const float*)d_in[8];
  const float* cs    = (const float*)d_in[9];

  float* out  = (float*)d_out;                        // [B, L, DK]
  float* attn = out + (size_t)B_ * L_ * DK_;          // [B, L, L]

  // Workspace carve-up (all bf16): 3x Wt (256 KB) + Qb,Kb (2 MB) + Vt (2 MB)
  char* ws = (char*)d_ws;
  bf16_t* Wqt = (bf16_t*)ws;  ws += (size_t)D_ * DK_ * sizeof(bf16_t);
  bf16_t* Wkt = (bf16_t*)ws;  ws += (size_t)D_ * DK_ * sizeof(bf16_t);
  bf16_t* Wvt = (bf16_t*)ws;  ws += (size_t)D_ * DK_ * sizeof(bf16_t);
  bf16_t* Qb  = (bf16_t*)ws;  ws += (size_t)B_ * L_ * DK_ * sizeof(bf16_t);
  bf16_t* Kb  = (bf16_t*)ws;  ws += (size_t)B_ * L_ * DK_ * sizeof(bf16_t);
  bf16_t* Vtp = (bf16_t*)ws;  ws += (size_t)B_ * L_ * DK_ * sizeof(bf16_t);

  // 1) transpose+convert the three weight matrices
  wtrans_kernel<<<(D_ * DK_) / 256, 256, 0, stream>>>(Wq, Wqt);
  wtrans_kernel<<<(D_ * DK_) / 256, 256, 0, stream>>>(Wk, Wkt);
  wtrans_kernel<<<(D_ * DK_) / 256, 256, 0, stream>>>(Wv, Wvt);

  // 2) projections (WMMA bf16, f32 accumulate)
  proj_kernel<<<(B_ * L_) / 16, 256, 0, stream>>>(query, Wqt, Qb, 0);
  proj_kernel<<<(B_ * L_) / 16, 256, 0, stream>>>(key,   Wkt, Kb, 0);
  proj_kernel<<<(B_ * L_) / 16, 256, 0, stream>>>(value, Wvt, Vtp, 1);

  // 3) raw scores -> attn region of d_out
  scores_kernel<<<dim3(L_ / 16, B_), 256, 0, stream>>>(Qb, Kb, mask, nph, nw, cs, attn);

  // 4) row softmax in place
  softmax_kernel<<<B_ * L_, 256, 0, stream>>>(attn);

  // 5) out = attn @ V with async-LDS staged attn tiles
  av_kernel<<<dim3(L_ / 16, B_), 256, 0, stream>>>(attn, Vtp, out);
}
